// EntitySlotAttnProcessor_29996051595794
// MI455X (gfx1250) — matrix operand order repaired and checked by reading, not compiled, not run on hardware
//
#include <hip/hip_runtime.h>
#include <cstdint>
#include <cstddef>

// Problem constants (match reference)
#define Bc   8
#define Sc   4096
#define Dc   640
#define Tc   77
#define DEc  768
#define Hc   10
#define HDc  64
#define Rc   64
#define TPc  96          // T padded to multiple of 32 for WMMA K
#define ATT_SCALE 0.125f // HD^-0.5

typedef _Float16 v16h __attribute__((ext_vector_type(16)));
typedef float    v8f  __attribute__((ext_vector_type(8)));

static __device__ __forceinline__ v8f wmma_f32_16x16x32_f16(v16h a, v16h b, v8f c) {
  return __builtin_amdgcn_wmma_f32_16x16x32_f16(false, a, false, b, (short)0, c,
                                                false, false);
}

// Async global->LDS copy, 16 bytes per lane (ASYNCcnt-tracked, no VGPR data).
// lds_off: LDS byte address for this lane; g: per-lane global source address.
static __device__ __forceinline__ void async_g2l_b128(uint32_t lds_off, const void* g) {
  asm volatile("global_load_async_to_lds_b128 %0, %1, off"
               :: "v"(lds_off), "v"(g)
               : "memory");
}

static __device__ __forceinline__ void wait_asynccnt0() {
  asm volatile("s_wait_asynccnt 0x0" ::: "memory");
}

// ---------------------------------------------------------------------------
// Generic GEMM: C[M,N] = A[M,K] * W[N,K]^T (+bias[N]) (optional SiLU)
// fp32 A/W in, f16 WMMA compute, fp32 C and/or f16 C16 out (either nullable).
// N multiple of 64, K multiple of 32. 256 thr = 8 waves, tile 128x64.
// ---------------------------------------------------------------------------
__global__ __launch_bounds__(256) void wmma_gemm_nt(
    const float* __restrict__ A, const float* __restrict__ W,
    const float* __restrict__ bias, float* __restrict__ C,
    _Float16* __restrict__ C16, int M, int N, int K, int act) {
  __shared__ __align__(16) _Float16 sA[128][40];  // 128 x 32 (+8 pad)
  __shared__ __align__(16) _Float16 sW[64][40];   // 64 x 32 (+8 pad)

  const int tid  = threadIdx.x;
  const int lane = tid & 31;
  const int wave = tid >> 5;
  const int lg   = lane & 15;
  const int kb   = (lane >> 4) << 3;   // 0 or 8 (K sub-offset per half-wave)
  const int m0   = blockIdx.x * 128;
  const int n0   = blockIdx.y * 64;

  v8f acc[4] = {};

  for (int k0 = 0; k0 < K; k0 += 32) {
    // ---- stage A tile: 128 rows x 32 cols, 16 elems/thread ----
    {
      const int row  = tid >> 1;
      const int c0   = (tid & 1) << 4;
      const int grow = m0 + row;
      float4 f0, f1, f2, f3;
      if (grow < M) {
        const float* p = A + (size_t)grow * K + k0 + c0;
        f0 = *(const float4*)(p + 0);
        f1 = *(const float4*)(p + 4);
        f2 = *(const float4*)(p + 8);
        f3 = *(const float4*)(p + 12);
        if (k0 + 32 < K) __builtin_prefetch(p + 32, 0, 1);  // global_prefetch
      } else {
        f0 = make_float4(0.f, 0.f, 0.f, 0.f); f1 = f0; f2 = f0; f3 = f0;
      }
      _Float16* d = &sA[row][c0];
      d[0]  = (_Float16)f0.x; d[1]  = (_Float16)f0.y;
      d[2]  = (_Float16)f0.z; d[3]  = (_Float16)f0.w;
      d[4]  = (_Float16)f1.x; d[5]  = (_Float16)f1.y;
      d[6]  = (_Float16)f1.z; d[7]  = (_Float16)f1.w;
      d[8]  = (_Float16)f2.x; d[9]  = (_Float16)f2.y;
      d[10] = (_Float16)f2.z; d[11] = (_Float16)f2.w;
      d[12] = (_Float16)f3.x; d[13] = (_Float16)f3.y;
      d[14] = (_Float16)f3.z; d[15] = (_Float16)f3.w;
    }
    // ---- stage W tile: 64 rows x 32 cols, 8 elems/thread ----
    {
      const int n  = tid >> 2;
      const int c0 = (tid & 3) << 3;
      const float* p = W + (size_t)(n0 + n) * K + k0 + c0;
      float4 f0 = *(const float4*)(p + 0);
      float4 f1 = *(const float4*)(p + 4);
      if (k0 + 32 < K) __builtin_prefetch(p + 32, 0, 1);
      _Float16* d = &sW[n][c0];
      d[0] = (_Float16)f0.x; d[1] = (_Float16)f0.y;
      d[2] = (_Float16)f0.z; d[3] = (_Float16)f0.w;
      d[4] = (_Float16)f1.x; d[5] = (_Float16)f1.y;
      d[6] = (_Float16)f1.z; d[7] = (_Float16)f1.w;
    }
    __syncthreads();

    // ---- A fragment (16x32 f16, wave-private rows) ----
    v16h af;
    const int ar = (wave << 4) + lg;
    #pragma unroll
    for (int i = 0; i < 8; ++i) {
      af[i]     = sA[ar][kb + i];
      af[8 + i] = sA[ar][16 + kb + i];
    }
    // ---- 4 WMMAs along N ----
    #pragma unroll
    for (int nt = 0; nt < 4; ++nt) {
      v16h bf;
      const int n = (nt << 4) + lg;
      #pragma unroll
      for (int i = 0; i < 8; ++i) {
        bf[i]     = sW[n][kb + i];
        bf[8 + i] = sW[n][16 + kb + i];
      }
      acc[nt] = wmma_f32_16x16x32_f16(af, bf, acc[nt]);
    }
    __syncthreads();
  }

  // ---- epilogue: bias / SiLU / store (C-matrix layout) ----
  const int rbase = m0 + (wave << 4) + ((lane >> 4) << 3);
  #pragma unroll
  for (int nt = 0; nt < 4; ++nt) {
    const int col = n0 + (nt << 4) + lg;
    const float bv = bias ? bias[col] : 0.0f;
    #pragma unroll
    for (int j = 0; j < 8; ++j) {
      const int row = rbase + j;
      if (row < M) {
        float c = acc[nt][j] + bv;
        if (act) c = c / (1.0f + __expf(-c));   // SiLU
        if (C)   C[(size_t)row * N + col] = c;
        if (C16) C16[(size_t)row * N + col] = (_Float16)c;
      }
    }
  }
}

// ---------------------------------------------------------------------------
// Fused attention: per (b,h, 128-query tile):
//   F_g  = softmax(q k^T * s) v        (WMMA, T padded 77->96)
//   F0/F1 = entity attention over 8 gathered tokens (fp32 VALU)
// q/k/v are f16 in global; K/V head slices staged to LDS via async-to-LDS.
// ---------------------------------------------------------------------------
__global__ __launch_bounds__(256) void attn_kernel(
    const _Float16* __restrict__ qf, const _Float16* __restrict__ kf,
    const _Float16* __restrict__ vf, const int* __restrict__ idx0,
    const int* __restrict__ idx1, float* __restrict__ Fg,
    float* __restrict__ F0, float* __restrict__ F1) {
  __shared__ __align__(16) _Float16 sK[TPc][HDc + 8];
  __shared__ __align__(16) _Float16 sV[TPc][HDc + 8];
  __shared__ __align__(16) _Float16 sP[128][TPc + 8];

  const int tid  = threadIdx.x;
  const int lane = tid & 31;
  const int wave = tid >> 5;
  const int lg   = lane & 15;
  const int kb   = (lane >> 4) << 3;
  const int b    = blockIdx.x / Hc;
  const int h    = blockIdx.x % Hc;
  const int s0   = blockIdx.y * 128;

  // ---- zero the pad rows [Tc, TPc) : 19 rows x 8 segs of 8 halfs ----
  for (int i = tid; i < (TPc - Tc) * 8; i += 256) {
    const int r = Tc + (i >> 3);
    const int c = (i & 7) << 3;
    *(float4*)&sK[r][c] = make_float4(0.f, 0.f, 0.f, 0.f);
    *(float4*)&sV[r][c] = make_float4(0.f, 0.f, 0.f, 0.f);
  }

  // ---- async global->LDS stage of K/V head slices (77 rows x 128B) ----
  {
    const _Float16* kb0 = kf + (size_t)b * Tc * Dc + h * HDc;
    const _Float16* vb0 = vf + (size_t)b * Tc * Dc + h * HDc;
    for (int i = tid; i < Tc * 8; i += 256) {
      const int r   = i >> 3;
      const int seg = (i & 7) << 3;  // 8 halfs = 16 bytes per chunk
      async_g2l_b128((uint32_t)(uintptr_t)&sK[r][seg], kb0 + (size_t)r * Dc + seg);
      async_g2l_b128((uint32_t)(uintptr_t)&sV[r][seg], vb0 + (size_t)r * Dc + seg);
    }
  }
  wait_asynccnt0();
  __syncthreads();

  // ---- scores: 6 N-tiles (96 cols) x 2 K-chunks over HD=64 ----
  v8f acc[6] = {};
  const int mrow = s0 + (wave << 4) + lg;
  const _Float16* qrow = qf + ((size_t)(b * Sc + mrow)) * Dc + h * HDc;
  #pragma unroll
  for (int kc = 0; kc < HDc; kc += 32) {
    v16h af;
    #pragma unroll
    for (int i = 0; i < 8; ++i) {
      af[i]     = qrow[kc + kb + i];
      af[8 + i] = qrow[kc + kb + 16 + i];
    }
    #pragma unroll
    for (int nt = 0; nt < 6; ++nt) {
      v16h bf;
      const int tcol = (nt << 4) + lg;
      #pragma unroll
      for (int i = 0; i < 8; ++i) {
        bf[i]     = sK[tcol][kc + kb + i];
        bf[8 + i] = sK[tcol][kc + kb + 16 + i];
      }
      acc[nt] = wmma_f32_16x16x32_f16(af, bf, acc[nt]);
    }
  }

  // ---- register softmax across 96 columns (16-lane half-wave reduce) ----
  #pragma unroll
  for (int j = 0; j < 8; ++j) {
    float m = -1e30f;
    #pragma unroll
    for (int nt = 0; nt < 6; ++nt) {
      const int col = (nt << 4) + lg;
      float vsc = (col < Tc) ? acc[nt][j] * ATT_SCALE : -1e30f;
      acc[nt][j] = vsc;
      m = fmaxf(m, vsc);
    }
    #pragma unroll
    for (int off = 1; off < 16; off <<= 1)
      m = fmaxf(m, __shfl_xor(m, off, 32));
    float ssum = 0.0f;
    #pragma unroll
    for (int nt = 0; nt < 6; ++nt) {
      float e = __expf(acc[nt][j] - m);
      acc[nt][j] = e;
      ssum += e;
    }
    #pragma unroll
    for (int off = 1; off < 16; off <<= 1)
      ssum += __shfl_xor(ssum, off, 32);
    const float inv = 1.0f / ssum;
    const int prow = (wave << 4) + j + ((lane >> 4) << 3);
    #pragma unroll
    for (int nt = 0; nt < 6; ++nt)
      sP[prow][(nt << 4) + lg] = (_Float16)(acc[nt][j] * inv);
  }
  __syncthreads();

  // ---- out = P(16x96) x V(96x64): 4 N-tiles x 3 K-chunks ----
  v8f accv[4] = {};
  #pragma unroll
  for (int kc = 0; kc < TPc; kc += 32) {
    v16h af;
    const int pr = (wave << 4) + lg;
    #pragma unroll
    for (int i = 0; i < 8; ++i) {
      af[i]     = sP[pr][kc + kb + i];
      af[8 + i] = sP[pr][kc + kb + 16 + i];
    }
    #pragma unroll
    for (int ntv = 0; ntv < 4; ++ntv) {
      v16h bf;
      const int n = (ntv << 4) + lg;
      #pragma unroll
      for (int i = 0; i < 8; ++i) {
        bf[i]     = sV[kc + kb + i][n];
        bf[8 + i] = sV[kc + kb + 16 + i][n];
      }
      accv[ntv] = wmma_f32_16x16x32_f16(af, bf, accv[ntv]);
    }
  }
  {
    const int rbase = s0 + (wave << 4) + ((lane >> 4) << 3);
    #pragma unroll
    for (int ntv = 0; ntv < 4; ++ntv) {
      const int hd = (ntv << 4) + lg;
      #pragma unroll
      for (int j = 0; j < 8; ++j) {
        const int row = rbase + j;
        Fg[((size_t)(b * Sc + row)) * Dc + h * HDc + hd] = accv[ntv][j];
      }
    }
  }
  __syncthreads();

  // ---- entity attention: one query row per thread (8 keys, fp32) ----
  {
    const int ent = tid >> 7;            // 0 or 1
    const int r   = tid & 127;
    const int srow = s0 + r;
    const int* ip = ent ? idx1 : idx0;
    int tj[8];
    #pragma unroll
    for (int j = 0; j < 8; ++j) tj[j] = ip[j];

    const size_t base = ((size_t)(b * Sc + srow)) * Dc + h * HDc;
    float qv[HDc];
    #pragma unroll
    for (int d = 0; d < HDc; ++d) qv[d] = (float)qf[base + d];

    float sc[8];
    float m = -1e30f;
    #pragma unroll
    for (int j = 0; j < 8; ++j) {
      float dot = 0.0f;
      const int t = tj[j];
      #pragma unroll
      for (int d = 0; d < HDc; ++d) dot += qv[d] * (float)sK[t][d];
      sc[j] = dot * ATT_SCALE;
      m = fmaxf(m, sc[j]);
    }
    float ssum = 0.0f;
    #pragma unroll
    for (int j = 0; j < 8; ++j) { sc[j] = __expf(sc[j] - m); ssum += sc[j]; }
    const float inv = 1.0f / ssum;
    #pragma unroll
    for (int j = 0; j < 8; ++j) sc[j] *= inv;

    float* Fo = ent ? F1 : F0;
    #pragma unroll
    for (int d = 0; d < HDc; ++d) {
      float o = 0.0f;
      #pragma unroll
      for (int j = 0; j < 8; ++j) o += sc[j] * (float)sV[tj[j]][d];
      Fo[base + d] = o;
    }
  }
}

// ---------------------------------------------------------------------------
// LayerNorm over D=640, one row per 128-thread block
// ---------------------------------------------------------------------------
__global__ __launch_bounds__(128) void ln_kernel(
    const float* __restrict__ x, const float* __restrict__ g,
    const float* __restrict__ bt, float* __restrict__ y) {
  __shared__ float red[128];
  const int row = blockIdx.x;
  const int tid = threadIdx.x;
  const float* xr = x + (size_t)row * Dc;
  float xv[5];
  float s = 0.0f;
  #pragma unroll
  for (int i = 0; i < 5; ++i) { xv[i] = xr[tid + i * 128]; s += xv[i]; }
  red[tid] = s;
  __syncthreads();
  for (int o = 64; o > 0; o >>= 1) {
    if (tid < o) red[tid] += red[tid + o];
    __syncthreads();
  }
  const float mean = red[0] * (1.0f / Dc);
  __syncthreads();
  float v = 0.0f;
  #pragma unroll
  for (int i = 0; i < 5; ++i) { float d = xv[i] - mean; v += d * d; }
  red[tid] = v;
  __syncthreads();
  for (int o = 64; o > 0; o >>= 1) {
    if (tid < o) red[tid] += red[tid + o];
    __syncthreads();
  }
  const float rstd = rsqrtf(red[0] * (1.0f / Dc) + 1e-5f);
  #pragma unroll
  for (int i = 0; i < 5; ++i) {
    const int c = tid + i * 128;
    y[(size_t)row * Dc + c] = (xv[i] - mean) * rstd * g[c] + bt[c];
  }
}

// ---------------------------------------------------------------------------
// blended = sig(raw) * ((F0r+U0) + (F1r+U1) + Fg)/3 + (1-sig(raw)) * Fg
// ---------------------------------------------------------------------------
__global__ void blend_kernel(const float* __restrict__ F0r,
                             const float* __restrict__ U0,
                             const float* __restrict__ F1r,
                             const float* __restrict__ U1,
                             const float* __restrict__ Fg,
                             const float* __restrict__ raw,
                             float* __restrict__ out, int n) {
  const int i = blockIdx.x * blockDim.x + threadIdx.x;
  if (i >= n) return;
  const float blend = 1.0f / (1.0f + __expf(-raw[0]));
  const float f0 = F0r[i] + U0[i];
  const float f1 = F1r[i] + U1[i];
  const float fg = Fg[i];
  const float comp = (f0 + f1 + fg) * (1.0f / 3.0f);
  out[i] = blend * comp + (1.0f - blend) * fg;
}

// ---------------------------------------------------------------------------
extern "C" void kernel_launch(void* const* d_in, const int* in_sizes, int n_in,
                              void* d_out, int out_size, void* d_ws, size_t ws_size,
                              hipStream_t stream) {
  const float* hs    = (const float*)d_in[0];
  const float* enc   = (const float*)d_in[1];
  const float* Wq    = (const float*)d_in[2];
  const float* Wk    = (const float*)d_in[3];
  const float* Wv    = (const float*)d_in[4];
  const float* Wo    = (const float*)d_in[5];
  const float* bo    = (const float*)d_in[6];
  const float* ln0g  = (const float*)d_in[7];
  const float* ln0b  = (const float*)d_in[8];
  const float* d0W   = (const float*)d_in[9];
  const float* d0b   = (const float*)d_in[10];
  const float* u0W   = (const float*)d_in[11];
  const float* u0b   = (const float*)d_in[12];
  const float* ln1g  = (const float*)d_in[13];
  const float* ln1b  = (const float*)d_in[14];
  const float* d1W   = (const float*)d_in[15];
  const float* d1b   = (const float*)d_in[16];
  const float* u1W   = (const float*)d_in[17];
  const float* u1b   = (const float*)d_in[18];
  const float* raw   = (const float*)d_in[19];
  const int*   idx0  = (const int*)d_in[20];
  const int*   idx1  = (const int*)d_in[21];
  float* out = (float*)d_out;

  // workspace carve-up
  char* ws = (char*)d_ws;
  const size_t QS = (size_t)Bc * Sc * Dc;       // 20,971,520 elems
  const size_t HS = (size_t)Bc * Sc * Rc;       //  2,097,152 elems
  const size_t KS = (size_t)Bc * Tc * Dc;       //    394,240 elems
  float* blended = (float*)ws; ws += QS * 4;
  float* Fg      = (float*)ws; ws += QS * 4;
  float* F0      = (float*)ws; ws += QS * 4;
  float* F1      = (float*)ws; ws += QS * 4;
  float* tmp0    = (float*)ws; ws += QS * 4;
  float* tmp1    = (float*)ws; ws += QS * 4;
  float* Hb      = (float*)ws; ws += HS * 4;
  _Float16* qf16 = (_Float16*)ws; ws += QS * 2;
  _Float16* kf16 = (_Float16*)ws; ws += KS * 2;
  _Float16* vf16 = (_Float16*)ws; ws += KS * 2;

  const dim3 blk(256);
  const int MS = Bc * Sc;    // 32768
  const int MT = Bc * Tc;    // 616

  auto gemm = [&](const float* A, const float* W, const float* bias, float* C,
                  _Float16* C16, int M, int N, int K, int act) {
    dim3 grid((M + 127) / 128, N / 64);
    hipLaunchKernelGGL(wmma_gemm_nt, grid, blk, 0, stream,
                       A, W, bias, C, C16, M, N, K, act);
  };

  // projections -> f16 outputs for the attention stage
  gemm(hs,  Wq, nullptr, nullptr, qf16, MS, Dc, Dc,  0);
  gemm(enc, Wk, nullptr, nullptr, kf16, MT, Dc, DEc, 0);
  gemm(enc, Wv, nullptr, nullptr, vf16, MT, Dc, DEc, 0);

  // fused global + entity attention
  hipLaunchKernelGGL(attn_kernel, dim3(Bc * Hc, Sc / 128), blk, 0, stream,
                     qf16, kf16, vf16, idx0, idx1, Fg, F0, F1);

  // adapter 0: LN -> down(+SiLU) -> up(+bias)
  hipLaunchKernelGGL(ln_kernel, dim3(MS), dim3(128), 0, stream, F0, ln0g, ln0b, tmp0);
  gemm(tmp0, d0W, d0b, Hb,   nullptr, MS, Rc, Dc, 1);
  gemm(Hb,   u0W, u0b, tmp0, nullptr, MS, Dc, Rc, 0);

  // adapter 1
  hipLaunchKernelGGL(ln_kernel, dim3(MS), dim3(128), 0, stream, F1, ln1g, ln1b, tmp1);
  gemm(tmp1, d1W, d1b, Hb,   nullptr, MS, Rc, Dc, 1);
  gemm(Hb,   u1W, u1b, tmp1, nullptr, MS, Dc, Rc, 0);

  // compose + blend
  const int n = (int)QS;
  hipLaunchKernelGGL(blend_kernel, dim3((n + 255) / 256), dim3(256), 0, stream,
                     F0, tmp0, F1, tmp1, Fg, raw, blended, n);

  // output projection
  gemm(blended, Wo, bo, out, nullptr, MS, Dc, Dc, 0);
}
